// Qfunction_25632364822817
// MI455X (gfx1250) — matrix operand order repaired and checked by reading, not compiled, not run on hardware
//
#include <hip/hip_runtime.h>
#include <hip/hip_bf16.h>
#include <stdint.h>

// ---------------------------------------------------------------------------
// Types for CDNA5 WMMA (wave32, 16x16x32 bf16 -> f32)
// ---------------------------------------------------------------------------
typedef __attribute__((ext_vector_type(16))) __bf16 v16bf;
typedef __attribute__((ext_vector_type(8)))  float  v8f;

union AFrag { unsigned int u[8]; v16bf v; };
union BFrag { unsigned int u[8]; v16bf v; };

#define HDIM 256
#define FIN  128

__device__ __forceinline__ unsigned short f2bf_rne(float f) {
    unsigned int u = __float_as_uint(f);
    unsigned int r = (u + 0x7FFFu + ((u >> 16) & 1u)) >> 16;
    return (unsigned short)r;
}

// ---------------------------------------------------------------------------
// Utility fill
// ---------------------------------------------------------------------------
__global__ void fill_f32_kernel(float* __restrict__ p, float v, int n) {
    int i = blockIdx.x * blockDim.x + threadIdx.x;
    if (i < n) p[i] = v;
}

// ---------------------------------------------------------------------------
// f32 -> bf16 cast (RNE)
// ---------------------------------------------------------------------------
__global__ void cast_bf16_kernel(const float* __restrict__ src,
                                 unsigned short* __restrict__ dst, int n) {
    int i = blockIdx.x * blockDim.x + threadIdx.x;
    if (i < n) dst[i] = f2bf_rne(src[i]);
}

// ---------------------------------------------------------------------------
// Degree accumulation: deg[col] += 1 for each edge (self-loop preseeded = 1)
// ---------------------------------------------------------------------------
__global__ void degree_kernel(const long long* __restrict__ ei, float* __restrict__ deg,
                              int E) {
    int e = blockIdx.x * blockDim.x + threadIdx.x;
    if (e < E) {
        int c = (int)ei[(size_t)E + e];   // edge_index[1][e] = aggregation target
        atomicAdd(&deg[c], 1.0f);
    }
}

__global__ void rsqrt_kernel(float* __restrict__ deg, int n) {
    int i = blockIdx.x * blockDim.x + threadIdx.x;
    if (i < n) {
        float d = deg[i];
        deg[i] = (d > 0.0f) ? rsqrtf(d) : 0.0f;
    }
}

// ---------------------------------------------------------------------------
// Pack W (K x 256, row-major f32) into per-lane WMMA B-fragment layout:
//   out[((kstep*16 + ntile)*32 + lane)*8 + i] = bf16 pair {K=kb+2i, K=kb+2i+1}
//   with n = lane%16 (column within tile), kb = (lane/16)*16.
// ---------------------------------------------------------------------------
__global__ void packW_kernel(const float* __restrict__ W, unsigned int* __restrict__ out,
                             int K) {
    int idx = blockIdx.x * blockDim.x + threadIdx.x;     // (K/32)*16*32 fragments
    int total = (K >> 5) * 16 * 32;
    if (idx >= total) return;
    int lane = idx & 31;
    int nt   = (idx >> 5) & 15;
    int ks   = idx >> 9;
    int n    = lane & 15;
    int kb   = (lane >> 4) * 16;
    int col  = nt * 16 + n;
    unsigned int* o = out + (size_t)idx * 8;
#pragma unroll
    for (int i = 0; i < 8; ++i) {
        int k0 = ks * 32 + kb + 2 * i;
        unsigned short lo = f2bf_rne(W[(size_t)k0 * HDIM + col]);
        unsigned short hi = f2bf_rne(W[(size_t)(k0 + 1) * HDIM + col]);
        o[i] = (unsigned int)lo | ((unsigned int)hi << 16);
    }
}

// ---------------------------------------------------------------------------
// GEMM: C[M x 256] = A[M x K] (bf16 row-major) * Wpacked
// 256 threads = 8 waves/block; wave computes a 16 x 64 strip (4 wmma tiles).
// The block's packed-B slab (ksteps x 4 tiles x 1KB) is identical for all 8
// waves, so it is staged ONCE into LDS via global_load_async_to_lds_b128,
// synchronized with s_wait_asynccnt + one barrier, then consumed by ds reads.
// ---------------------------------------------------------------------------
__global__ void gemm_bf16_wmma_kernel(const unsigned short* __restrict__ A,
                                      const unsigned int* __restrict__ Bp,
                                      float* __restrict__ C, int M, int K) {
    __shared__ unsigned int smemB[8192];        // up to 8 ksteps * 1024 dwords = 32KB

    const int tid  = threadIdx.x;
    const int wave = tid >> 5;
    const int lane = tid & 31;
    const int row0 = (blockIdx.x * 8 + wave) * 16;
    const int nt0  = blockIdx.y * 4;
    const int ksteps = K >> 5;

    // ---- Cooperative async stage of packed B into LDS (all 256 threads) ----
    {
        unsigned lds0 = (unsigned)(uintptr_t)&smemB[0];
        unsigned voff = (unsigned)tid * 16u;    // 16 bytes per thread per kstep
        for (int ks = 0; ks < ksteps; ++ks) {
            unsigned ldsa = lds0 + (unsigned)(ks * 4096 + tid * 16);
            const unsigned int* sa = Bp + (size_t)(ks * 16 + nt0) * 256;
            asm volatile("global_load_async_to_lds_b128 %0, %1, %2"
                         :: "v"(ldsa), "v"(voff), "s"(sa)
                         : "memory");
        }
        asm volatile("s_wait_asynccnt 0" ::: "memory");
    }
    __syncthreads();   // all waves' async LDS writes visible

    if (row0 < M) {    // wave-uniform; EXEC all-ones inside (WMMA requirement)
        v8f acc0 = {}, acc1 = {}, acc2 = {}, acc3 = {};

        const int m     = lane & 15;
        const int khalf = (lane >> 4) * 16;
        const unsigned short* arow = A + (size_t)(row0 + m) * K;

        for (int ks = 0; ks < ksteps; ++ks) {
            int kbase = ks * 32;
            AFrag a;
#pragma unroll
            for (int j = 0; j < 4; ++j)
                a.u[j] = *(const unsigned int*)(arow + kbase + khalf + 2 * j);
#pragma unroll
            for (int j = 0; j < 4; ++j)
                a.u[4 + j] = *(const unsigned int*)(arow + kbase + khalf + 8 + 2 * j);

            const unsigned int* bl = &smemB[(unsigned)(ks * 1024 + lane * 8)];
            BFrag b0, b1, b2, b3;
#pragma unroll
            for (int i = 0; i < 8; ++i) b0.u[i] = bl[i];
#pragma unroll
            for (int i = 0; i < 8; ++i) b1.u[i] = bl[256 + i];
#pragma unroll
            for (int i = 0; i < 8; ++i) b2.u[i] = bl[512 + i];
#pragma unroll
            for (int i = 0; i < 8; ++i) b3.u[i] = bl[768 + i];

            acc0 = __builtin_amdgcn_wmma_f32_16x16x32_bf16(false, a.v, false, b0.v,
                                                           (short)0, acc0, false, false);
            acc1 = __builtin_amdgcn_wmma_f32_16x16x32_bf16(false, a.v, false, b1.v,
                                                           (short)0, acc1, false, false);
            acc2 = __builtin_amdgcn_wmma_f32_16x16x32_bf16(false, a.v, false, b2.v,
                                                           (short)0, acc2, false, false);
            acc3 = __builtin_amdgcn_wmma_f32_16x16x32_bf16(false, a.v, false, b3.v,
                                                           (short)0, acc3, false, false);
        }

        // C/D layout: vgpr r -> row = r + (lane/16)*8, col = ntile*16 + lane%16
        const int nlo = lane & 15;
        const int mhi = (lane >> 4) * 8;
        v8f accs[4] = {acc0, acc1, acc2, acc3};
#pragma unroll
        for (int t = 0; t < 4; ++t) {
            int col = (nt0 + t) * 16 + nlo;
#pragma unroll
            for (int r = 0; r < 8; ++r) {
                C[(size_t)(row0 + mhi + r) * HDIM + col] = accs[t][r];
            }
        }
    }
}

// ---------------------------------------------------------------------------
// Edge scatter: out[col] += xw[row] * dinv[row]*dinv[col]; one wave per edge
// (edges E.. E+N-1 are the self-loops). 8 floats per lane, fully coalesced.
// ---------------------------------------------------------------------------
__global__ void scatter_kernel(const float* __restrict__ xw,
                               const long long* __restrict__ ei,
                               const float* __restrict__ dinv,
                               float* __restrict__ out, int E, int N) {
    int gw   = (int)((blockIdx.x * (size_t)blockDim.x + threadIdx.x) >> 5);
    int lane = threadIdx.x & 31;
    int total = E + N;
    if (gw >= total) return;
    int r, c;
    if (gw < E) {
        r = (int)ei[gw];
        c = (int)ei[(size_t)E + gw];
    } else {
        r = c = gw - E;
    }
    float norm = dinv[r] * dinv[c];
    const float* src = xw + (size_t)r * HDIM;
    float*       dst = out + (size_t)c * HDIM;
#pragma unroll
    for (int j = 0; j < 8; ++j) {
        int f = lane + 32 * j;
        atomicAdd(dst + f, src[f] * norm);
    }
}

// ---------------------------------------------------------------------------
// h = relu(scat + bias); optionally write f32 and/or bf16 outputs
// ---------------------------------------------------------------------------
__global__ void bias_relu_kernel(const float* __restrict__ scat,
                                 const float* __restrict__ bias,
                                 float* __restrict__ outF,
                                 unsigned short* __restrict__ outB, int total) {
    int i = blockIdx.x * blockDim.x + threadIdx.x;
    if (i >= total) return;
    int f = i & (HDIM - 1);
    float v = scat[i] + bias[f];
    v = fmaxf(v, 0.0f);
    if (outF) outF[i] = v;
    if (outB) outB[i] = f2bf_rne(v);
}

// ---------------------------------------------------------------------------
// Global pooling: per-feature sum + max (h >= 0 post-ReLU, so int atomicMax
// on float bits with 0-init is order-correct).
// ---------------------------------------------------------------------------
#define POOL_ROWS 100
__global__ void pool_kernel(const float* __restrict__ h, float* __restrict__ sumb,
                            float* __restrict__ maxb, int N) {
    int f  = threadIdx.x;                 // 256 threads = one per feature
    int r0 = blockIdx.x * POOL_ROWS;
    int r1 = min(r0 + POOL_ROWS, N);
    float s = 0.0f, m = 0.0f;
    for (int r = r0; r < r1; ++r) {
        float v = h[(size_t)r * HDIM + f];
        s += v;
        m = fmaxf(m, v);
    }
    atomicAdd(&sumb[f], s);
    atomicMax((int*)&maxb[f], __float_as_int(m));
}

// ---------------------------------------------------------------------------
// Final: pooled = [mean | max | sum] (768) -> out = pooled @ Wlin + blin  (5)
// ---------------------------------------------------------------------------
__global__ void final_kernel(const float* __restrict__ sumb,
                             const float* __restrict__ maxb,
                             const float* __restrict__ Wlin,
                             const float* __restrict__ blin,
                             float* __restrict__ out, int N, int A) {
    __shared__ float pooled[3 * HDIM];
    int t = threadIdx.x;                  // 256 threads
    float s = sumb[t];
    pooled[t]            = s / (float)N;
    pooled[HDIM + t]     = maxb[t];
    pooled[2 * HDIM + t] = s;
    __syncthreads();
    if (t < A) {
        float acc = blin[t];
        for (int k = 0; k < 3 * HDIM; ++k) acc += pooled[k] * Wlin[(size_t)k * A + t];
        out[t] = acc;
    }
}

// ---------------------------------------------------------------------------
// Host launcher
// ---------------------------------------------------------------------------
extern "C" void kernel_launch(void* const* d_in, const int* in_sizes, int n_in,
                              void* d_out, int out_size, void* d_ws, size_t ws_size,
                              hipStream_t stream) {
    const float*     x    = (const float*)d_in[0];
    const long long* ei   = (const long long*)d_in[1];   // int64 edge_index [2, E]
    const float*     W1   = (const float*)d_in[2];
    const float*     b1   = (const float*)d_in[3];
    const float*     W2   = (const float*)d_in[4];
    const float*     b2   = (const float*)d_in[5];
    const float*     Wlin = (const float*)d_in[6];
    const float*     blin = (const float*)d_in[7];
    float*           out  = (float*)d_out;

    const int Nn = in_sizes[0] / FIN;     // 50000
    const int E  = in_sizes[1] / 2;       // 800000
    const int A  = in_sizes[7];           // 5

    // Workspace carve-up (256B aligned)
    char* base = (char*)d_ws;
    size_t off = 0;
    auto carve = [&](size_t bytes) -> char* {
        char* p = base + off;
        off = (off + bytes + 255) & ~(size_t)255;
        return p;
    };
    float*          dinv = (float*)carve((size_t)Nn * 4);
    float*          xw   = (float*)carve((size_t)Nn * HDIM * 4);
    float*          hs   = (float*)carve((size_t)Nn * HDIM * 4);
    unsigned short* xb   = (unsigned short*)carve((size_t)Nn * FIN * 2);
    unsigned short* h1b  = (unsigned short*)carve((size_t)Nn * HDIM * 2);
    unsigned int*   wp   = (unsigned int*)carve((size_t)(HDIM / 32) * 16 * 32 * 8 * 4);
    float*          sumb = (float*)carve(HDIM * 4);
    float*          maxb = (float*)carve(HDIM * 4);

    const int T = 256;
    auto blocks = [](long long n, int t) { return (unsigned int)((n + t - 1) / t); };

    // Degree -> dinv (self-loop contributes 1.0 seed)
    fill_f32_kernel<<<blocks(Nn, T), T, 0, stream>>>(dinv, 1.0f, Nn);
    degree_kernel<<<blocks(E, T), T, 0, stream>>>(ei, dinv, E);
    rsqrt_kernel<<<blocks(Nn, T), T, 0, stream>>>(dinv, Nn);

    // x -> bf16
    cast_bf16_kernel<<<blocks((long long)Nn * FIN, T), T, 0, stream>>>(x, xb, Nn * FIN);

    // ---- Layer 1 ----
    packW_kernel<<<blocks((FIN / 32) * 16 * 32, T), T, 0, stream>>>(W1, wp, FIN);
    {
        dim3 g(blocks((long long)(Nn / 16), 8), HDIM / 64);
        gemm_bf16_wmma_kernel<<<g, T, 0, stream>>>(xb, wp, xw, Nn, FIN);
    }
    fill_f32_kernel<<<blocks((long long)Nn * HDIM, T), T, 0, stream>>>(hs, 0.0f, Nn * HDIM);
    scatter_kernel<<<blocks((long long)(E + Nn) * 32, T), T, 0, stream>>>(xw, ei, dinv, hs, E, Nn);
    bias_relu_kernel<<<blocks((long long)Nn * HDIM, T), T, 0, stream>>>(hs, b1, nullptr, h1b,
                                                                        Nn * HDIM);

    // ---- Layer 2 ----
    packW_kernel<<<blocks((HDIM / 32) * 16 * 32, T), T, 0, stream>>>(W2, wp, HDIM);
    {
        dim3 g(blocks((long long)(Nn / 16), 8), HDIM / 64);
        gemm_bf16_wmma_kernel<<<g, T, 0, stream>>>(h1b, wp, xw, Nn, HDIM);
    }
    fill_f32_kernel<<<blocks((long long)Nn * HDIM, T), T, 0, stream>>>(hs, 0.0f, Nn * HDIM);
    scatter_kernel<<<blocks((long long)(E + Nn) * 32, T), T, 0, stream>>>(xw, ei, dinv, hs, E, Nn);
    bias_relu_kernel<<<blocks((long long)Nn * HDIM, T), T, 0, stream>>>(hs, b2, hs, nullptr,
                                                                        Nn * HDIM);

    // ---- Pool + head ----
    fill_f32_kernel<<<1, T, 0, stream>>>(sumb, 0.0f, HDIM);
    fill_f32_kernel<<<1, T, 0, stream>>>(maxb, 0.0f, HDIM);
    pool_kernel<<<blocks(Nn, POOL_ROWS), T, 0, stream>>>(hs, sumb, maxb, Nn);
    final_kernel<<<1, T, 0, stream>>>(sumb, maxb, Wlin, blin, out, Nn, A);
}
